// GATLayer_66907000537647
// MI455X (gfx1250) — matrix operand order repaired and checked by reading, not compiled, not run on hardware
//
#include <hip/hip_runtime.h>
#include <hip/hip_bf16.h>

// ---------- CDNA5 WMMA types ----------
typedef __attribute__((ext_vector_type(16))) _Float16 v16h;
typedef __attribute__((ext_vector_type(8)))  _Float16 v8h;
typedef __attribute__((ext_vector_type(4)))  _Float16 v4h;
typedef __attribute__((ext_vector_type(8)))  float    v8f;

union AV { v16h v; v8h h[2]; };

#define N_NODES 8192
#define F_DIM   256
#define ALPHA   0.2f
#define NEG_BIG -9e15f

__device__ inline v8f wmma_f16(v16h a, v16h b, v8f c) {
    return __builtin_amdgcn_wmma_f32_16x16x32_f16(false, a, false, b,
                                                  (short)0, c, false, false);
}

// load 8 consecutive f32 and convert to v8h
__device__ inline v8h cvt8(const float* p) {
    float4 x = ((const float4*)p)[0];
    float4 y = ((const float4*)p)[1];
    v8h r;
    r[0] = (_Float16)x.x; r[1] = (_Float16)x.y;
    r[2] = (_Float16)x.z; r[3] = (_Float16)x.w;
    r[4] = (_Float16)y.x; r[5] = (_Float16)y.y;
    r[6] = (_Float16)y.z; r[7] = (_Float16)y.w;
    return r;
}

// ---------- Kernel 1: WT[n][k] = (f16) W[k][n] ----------
__global__ __launch_bounds__(256) void k_prep_wt(const float* __restrict__ W,
                                                 _Float16* __restrict__ WT) {
    int idx = blockIdx.x * 256 + threadIdx.x;     // idx = n*256 + k
    int n = idx >> 8, k = idx & 255;
    WT[idx] = (_Float16)W[k * F_DIM + n];
}

// ---------- Kernel 2: wh = h @ W  (f16 WMMA, f32 accum) ----------
// Block: 256 thr = 8 waves arranged 2(M) x 4(N). Block tile 32 x 256.
__global__ __launch_bounds__(256) void k_wh_gemm(const float* __restrict__ h,
                                                 const _Float16* __restrict__ WT,
                                                 _Float16* __restrict__ whr,   // [N][F] f16
                                                 _Float16* __restrict__ whT) { // [F][N] f16
    const int tid  = threadIdx.x;
    const int lane = tid & 31;
    const int wave = tid >> 5;
    const int wy = wave >> 2, wx = wave & 3;
    const int m0 = blockIdx.x * 32 + wy * 16;
    const int n0 = wx * 64;

    v8f acc[4];
#pragma unroll
    for (int t = 0; t < 4; ++t) acc[t] = (v8f)0.f;

    const int arow = m0 + (lane & 15);
    const int klo  = (lane & 16) ? 8 : 0;    // A-frag K split (16-bit layout)
    const int kbhi = (lane & 16) ? 8 : 0;    // B-frag K split

#pragma unroll 2
    for (int kc = 0; kc < F_DIM; kc += 32) {
        // preload all fragments first so the 8 b128 loads overlap
        AV bf[4];
#pragma unroll
        for (int nt = 0; nt < 4; ++nt) {
            const _Float16* bp = WT + (n0 + nt * 16 + (lane & 15)) * F_DIM + kc + kbhi;
            bf[nt].h[0] = *(const v8h*)(bp);
            bf[nt].h[1] = *(const v8h*)(bp + 16);
        }
        AV af;
        af.h[0] = cvt8(h + arow * F_DIM + kc + klo);
        af.h[1] = cvt8(h + arow * F_DIM + kc + klo + 16);
#pragma unroll
        for (int nt = 0; nt < 4; ++nt)
            acc[nt] = wmma_f16(af.v, bf[nt].v, acc[nt]);
    }

    // C/D layout: VGPR r -> M = r (+8 for lanes>=16), N = lane&15
    const int mo = (lane & 16) ? 8 : 0;
#pragma unroll
    for (int nt = 0; nt < 4; ++nt) {
        const int Nn = n0 + nt * 16 + (lane & 15);
#pragma unroll
        for (int r = 0; r < 8; ++r) {
            const int M = m0 + r + mo;
            _Float16 hv = (_Float16)acc[nt][r];
            whr[M * F_DIM + Nn] = hv;
            whT[Nn * N_NODES + M] = hv;
        }
    }
}

// ---------- Kernel 3: wh1[i] = wh[i,:]·a[:256], wh2[i] = wh[i,:]·a[256:] ----------
__global__ __launch_bounds__(256) void k_wh12(const _Float16* __restrict__ whr,
                                              const float* __restrict__ a,
                                              float* __restrict__ wh1,
                                              float* __restrict__ wh2) {
    const int lane = threadIdx.x & 31;
    const int wave = threadIdx.x >> 5;
    const int row  = blockIdx.x * 8 + wave;
    v8h hv = *(const v8h*)(whr + row * F_DIM + lane * 8);
    float s1 = 0.f, s2 = 0.f;
#pragma unroll
    for (int k = 0; k < 8; ++k) {
        float x = (float)hv[k];
        s1 += x * a[lane * 8 + k];
        s2 += x * a[F_DIM + lane * 8 + k];
    }
#pragma unroll
    for (int m = 16; m >= 1; m >>= 1) {
        s1 += __shfl_xor(s1, m, 32);
        s2 += __shfl_xor(s2, m, 32);
    }
    if (lane == 0) { wh1[row] = s1; wh2[row] = s2; }
}

// ---------- Kernel 4: fused leaky/mask/softmax + P@wh + ELU ----------
// One block = 16 output rows. 8 waves, wave owns 32 output features.
// Chunked J=64: 4 score elements / thread, 4 WMMAs / wave / chunk.
__global__ __launch_bounds__(256) void k_gat_attn(const int* __restrict__ adj,
                                                  const float* __restrict__ wh1,
                                                  const float* __restrict__ wh2,
                                                  const _Float16* __restrict__ whT,
                                                  float* __restrict__ out) {
    __shared__ __align__(16) float wh2s[N_NODES];
    __shared__ float wh1s[16];
    __shared__ __align__(16) _Float16 ptile[16 * 64];
    __shared__ float scale_s[16];
    __shared__ float l_s[16];

    const int tid  = threadIdx.x;
    const int lane = tid & 31;
    const int wave = tid >> 5;
    const int i0   = blockIdx.x * 16;

    // ---- async-stage wh2 (32 KB) into LDS: GLOBAL_LOAD_ASYNC_TO_LDS_B128 ----
#pragma unroll
    for (int it = 0; it < 8; ++it) {
        const int idx = (it * 256 + tid) * 4;
        unsigned lds_off = (unsigned)(size_t)&wh2s[idx];
        const float* src = wh2 + idx;
        asm volatile("global_load_async_to_lds_b128 %0, %1, off"
                     :: "v"(lds_off), "v"(src) : "memory");
    }
    if (tid < 16) wh1s[tid] = wh1[i0 + tid];
    asm volatile("s_wait_asynccnt 0x0" ::: "memory");
    __syncthreads();

    // score-tile mapping: 16 threads per row, 4 consecutive cols per thread;
    // each row lives entirely inside one 16-lane half-wave
    const int row = tid >> 4;            // 0..15
    const int c4  = (tid & 15) * 4;      // column quad
    const long arowbase = (long)(i0 + row) * N_NODES;
    const float b1 = wh1s[row];

    // online-softmax state, replicated across the 16 lanes of a row group
    float m_run = -INFINITY;
    float l_run = 0.f;

    // accumulators: wave owns feature cols [c0, c0+32)
    const int c0 = wave * 32;
    v8f acc0 = (v8f)0.f, acc1 = (v8f)0.f;

    const int klo   = (lane & 16) ? 8 : 0;    // A-frag half select
    const int kbhi  = (lane & 16) ? 8 : 0;    // B-frag half select
    const int mo    = (lane & 16) ? 8 : 0;    // C/D row offset
    const _Float16* bbase0 = whT + (long)(c0 + (lane & 15)) * N_NODES;
    const _Float16* bbase1 = bbase0 + 16L * N_NODES;

    for (int jc = 0; jc < N_NODES; jc += 64) {
        // ---- scores for 16x64 tile (4 per thread) ----
        int4 ad = *(const int4*)(adj + arowbase + jc + c4);
        __builtin_prefetch(adj + arowbase + jc + c4 + 512, 0, 0);
        float4 w2 = *(const float4*)&wh2s[jc + c4];
        float e0 = b1 + w2.x, e1 = b1 + w2.y, e2 = b1 + w2.z, e3 = b1 + w2.w;
        e0 = e0 > 0.f ? e0 : ALPHA * e0;
        e1 = e1 > 0.f ? e1 : ALPHA * e1;
        e2 = e2 > 0.f ? e2 : ALPHA * e2;
        e3 = e3 > 0.f ? e3 : ALPHA * e3;
        if (ad.x <= 0) e0 = NEG_BIG;
        if (ad.y <= 0) e1 = NEG_BIG;
        if (ad.z <= 0) e2 = NEG_BIG;
        if (ad.w <= 0) e3 = NEG_BIG;

        // ---- online softmax (butterfly within 16-lane row group) ----
        float mx = fmaxf(fmaxf(e0, e1), fmaxf(e2, e3));
#pragma unroll
        for (int m = 8; m >= 1; m >>= 1) mx = fmaxf(mx, __shfl_xor(mx, m, 32));
        float m_new = fmaxf(m_run, mx);
        float sc = __expf(m_run - m_new);
        float p0 = __expf(e0 - m_new);
        float p1 = __expf(e1 - m_new);
        float p2 = __expf(e2 - m_new);
        float p3 = __expf(e3 - m_new);
        float ps = (p0 + p1) + (p2 + p3);
#pragma unroll
        for (int m = 8; m >= 1; m >>= 1) ps += __shfl_xor(ps, m, 32);
        l_run = l_run * sc + ps;
        m_run = m_new;

        if ((tid & 15) == 0) scale_s[row] = sc;
        v4h pv;
        pv[0] = (_Float16)p0; pv[1] = (_Float16)p1;
        pv[2] = (_Float16)p2; pv[3] = (_Float16)p3;
        *(v4h*)&ptile[row * 64 + c4] = pv;
        __syncthreads();

        // ---- rescale accumulators by exp(m_old - m_new), once per 64 cols ----
#pragma unroll
        for (int r = 0; r < 8; ++r) {
            float sv = scale_s[r + mo];
            acc0[r] *= sv;
            acc1[r] *= sv;
        }

        // ---- A-frags (P tile from LDS): K = 0..31 and 32..63 ----
        AV af0, af1;
        {
            const _Float16* ap = ptile + (lane & 15) * 64 + klo;
            af0.h[0] = *(const v8h*)(ap);      af0.h[1] = *(const v8h*)(ap + 16);
            af1.h[0] = *(const v8h*)(ap + 32); af1.h[1] = *(const v8h*)(ap + 48);
        }
        // ---- B-frags (whT, per-lane K in A-frag table order) ----
        AV bf00, bf01, bf10, bf11;
        {
            const _Float16* p0b = bbase0 + jc + kbhi;
            const _Float16* p1b = bbase1 + jc + kbhi;
            bf00.h[0] = *(const v8h*)(p0b);      bf00.h[1] = *(const v8h*)(p0b + 16);
            bf01.h[0] = *(const v8h*)(p0b + 32); bf01.h[1] = *(const v8h*)(p0b + 48);
            bf10.h[0] = *(const v8h*)(p1b);      bf10.h[1] = *(const v8h*)(p1b + 16);
            bf11.h[0] = *(const v8h*)(p1b + 32); bf11.h[1] = *(const v8h*)(p1b + 48);
        }
        acc0 = wmma_f16(af0.v, bf00.v, acc0);
        acc0 = wmma_f16(af1.v, bf01.v, acc0);
        acc1 = wmma_f16(af0.v, bf10.v, acc1);
        acc1 = wmma_f16(af1.v, bf11.v, acc1);
        __syncthreads();
    }

    if ((tid & 15) == 0) l_s[row] = l_run;
    __syncthreads();

    // ---- epilogue: divide by l, ELU, store f32 ----
#pragma unroll
    for (int r = 0; r < 8; ++r) {
        const int M = r + mo;
        const float linv = 1.0f / l_s[M];
        float v0 = acc0[r] * linv;
        float v1 = acc1[r] * linv;
        v0 = v0 > 0.f ? v0 : (__expf(v0) - 1.0f);
        v1 = v1 > 0.f ? v1 : (__expf(v1) - 1.0f);
        out[(long)(i0 + M) * F_DIM + c0 + (lane & 15)]      = v0;
        out[(long)(i0 + M) * F_DIM + c0 + 16 + (lane & 15)] = v1;
    }
}

// ---------- launch ----------
extern "C" void kernel_launch(void* const* d_in, const int* in_sizes, int n_in,
                              void* d_out, int out_size, void* d_ws, size_t ws_size,
                              hipStream_t stream) {
    const float* h   = (const float*)d_in[0];
    const int*   adj = (const int*)d_in[1];
    const float* W   = (const float*)d_in[2];
    const float* a   = (const float*)d_in[3];
    float* out = (float*)d_out;

    char* ws = (char*)d_ws;
    _Float16* WT  = (_Float16*)(ws);                                   // 128 KB
    _Float16* whr = (_Float16*)(ws + (size_t)131072);                  // 4 MB
    _Float16* whT = (_Float16*)(ws + (size_t)131072 + 4194304);        // 4 MB
    float*    wh1 = (float*)   (ws + (size_t)131072 + 8388608);        // 32 KB
    float*    wh2 = (float*)   (ws + (size_t)131072 + 8388608 + 32768);// 32 KB

    k_prep_wt<<<F_DIM * F_DIM / 256, 256, 0, stream>>>(W, WT);
    k_wh_gemm<<<N_NODES / 32, 256, 0, stream>>>(h, WT, whr, whT);
    k_wh12<<<N_NODES / 8, 256, 0, stream>>>(whr, a, wh1, wh2);
    k_gat_attn<<<N_NODES / 16, 256, 0, stream>>>(adj, wh1, wh2, whT, out);
}
